// RGCN_node_49727131353506
// MI455X (gfx1250) — compile-verified, matrix operand order checked by reading, not run on hardware
//
#include <hip/hip_runtime.h>

#define N_NODESC 100000
#define N_EDGESC 3200000
#define F_INC    128
#define HIDC     16
#define N_CLS    32
#define N_RELSC  20
#define K2C      352   // 20*16 relation rows + 16 root rows + 16 zero pad
#define BN_EPSF  1e-5f

typedef _Float16 h8   __attribute__((ext_vector_type(8)));
typedef _Float16 v16h __attribute__((ext_vector_type(16)));
typedef float    v8f  __attribute__((ext_vector_type(8)));

// ---------------- elementwise f32 -> f16 convert ----------------
__global__ void k_cvt_f16(const float* __restrict__ x, _Float16* __restrict__ y, int n) {
  int i = blockIdx.x * blockDim.x + threadIdx.x;
  if (i < n) y[i] = (_Float16)x[i];
}

// ---------------- pack W1 (+root1 as rel 20) into WMMA B layout ----------------
// W1p[((r*4+kk)*32 + lane)*16 + j] ; lane holds column n=lane&15,
// element j -> K = kk*32 + (j<8 ? j : j+8) + (lane>=16 ? 8 : 0)
__global__ void k_pack_w1(const float* __restrict__ W1, const float* __restrict__ root1,
                          _Float16* __restrict__ W1p) {
  int idx = blockIdx.x * blockDim.x + threadIdx.x;
  if (idx >= (N_RELSC + 1) * 4 * 512) return;
  int j    = idx & 15;
  int lane = (idx >> 4) & 31;
  int kk   = (idx >> 9) & 3;
  int r    = idx >> 11;
  int n = lane & 15;
  int k = kk * 32 + ((j < 8) ? j : j + 8) + ((lane >= 16) ? 8 : 0);
  float v = (r < N_RELSC) ? W1[(r * F_INC + k) * HIDC + n] : root1[k * HIDC + n];
  W1p[idx] = (_Float16)v;
}

// ---------------- pack concatenated [W2 ; root2 ; 0] (352 x 32) into B layout ----
// two n-tiles (cols 0-15, 16-31), 11 k-steps of 32
__global__ void k_pack_w2(const float* __restrict__ W2, const float* __restrict__ root2,
                          _Float16* __restrict__ W2p) {
  int idx = blockIdx.x * blockDim.x + threadIdx.x;
  if (idx >= 2 * 11 * 512) return;
  int j    = idx & 15;
  int lane = (idx >> 4) & 31;
  int q    = idx >> 9;        // t*11 + kk
  int t    = q / 11;
  int kk   = q % 11;
  int col = t * 16 + (lane & 15);
  int k = kk * 32 + ((j < 8) ? j : j + 8) + ((lane >= 16) ? 8 : 0);
  float v;
  if (k < N_RELSC * HIDC)            v = W2[k * N_CLS + col];
  else if (k < N_RELSC * HIDC + 16)  v = root2[(k - N_RELSC * HIDC) * N_CLS + col];
  else                               v = 0.f;
  W2p[idx] = (_Float16)v;
}

// ---------------- layer-1 dense: H1[r] = Xh @ W1[r] (f16, f32 acc) --------------
// wave per 16-node tile; r==20 is the root term -> seeds out1 with x@root1 + b1
__global__ void k_gemm1(const _Float16* __restrict__ Xh, const _Float16* __restrict__ W1p,
                        const float* __restrict__ b1,
                        _Float16* __restrict__ H1, float* __restrict__ out1) {
  int wave  = blockIdx.x * (blockDim.x >> 5) + (threadIdx.x >> 5);
  int lane  = threadIdx.x & 31;
  int node0 = wave * 16;
  if (node0 >= N_NODESC) return;           // whole-wave exit: EXEC all-ones for WMMA
  int m  = lane & 15;
  int hi = lane >> 4;
  const _Float16* xrow = Xh + (size_t)(node0 + m) * F_INC;
  v16h a[4];
#pragma unroll
  for (int kk = 0; kk < 4; ++kk) {
    h8 lo = *(const h8*)(xrow + kk * 32 + hi * 8);
    h8 hh = *(const h8*)(xrow + kk * 32 + hi * 8 + 16);
    a[kk] = __builtin_shufflevector(lo, hh, 0,1,2,3,4,5,6,7,8,9,10,11,12,13,14,15);
  }
  float bias = b1[m];
  for (int r = 0; r <= N_RELSC; ++r) {
    v8f c = {};
#pragma unroll
    for (int kk = 0; kk < 4; ++kk) {
      v16h b = *(const v16h*)(W1p + (size_t)(((r * 4 + kk) * 32) + lane) * 16);
      c = __builtin_amdgcn_wmma_f32_16x16x32_f16(false, a[kk], false, b,
                                                 (short)0, c, false, false);
    }
    if (r < N_RELSC) {
      _Float16* dst = H1 + ((size_t)r * N_NODESC + node0 + hi * 8) * HIDC + m;
#pragma unroll
      for (int i = 0; i < 8; ++i) dst[(size_t)i * HIDC] = (_Float16)c[i];
    } else {
      float* dst = out1 + (size_t)(node0 + hi * 8) * HIDC + m;
#pragma unroll
      for (int i = 0; i < 8; ++i) dst[(size_t)i * HIDC] = c[i] + bias;
    }
  }
}

// ---------------- per-(rel,dst) in-degree + inverse --------------------------
__global__ void k_count(const int* __restrict__ ei, const int* __restrict__ et,
                        float* __restrict__ cnt) {
  int e = blockIdx.x * blockDim.x + threadIdx.x;
  if (e >= N_EDGESC) return;
  int d = ei[N_EDGESC + e];
  int r = et[e];
  atomicAdd(&cnt[(size_t)r * N_NODESC + d], 1.0f);
}
__global__ void k_invert(float* __restrict__ cnt, int n) {
  int i = blockIdx.x * blockDim.x + threadIdx.x;
  if (i < n) { float v = cnt[i]; cnt[i] = (v > 0.f) ? 1.f / v : 0.f; }
}

// ---------------- layer-1 edge scatter: out1[dst] += H1[r,src]/cnt -----------
__global__ void k_edge1(const int* __restrict__ ei, const int* __restrict__ et,
                        const _Float16* __restrict__ H1, const float* __restrict__ invc,
                        float* __restrict__ out1) {
  int e = blockIdx.x * blockDim.x + threadIdx.x;
  if (e >= N_EDGESC) return;
  int s = ei[e], d = ei[N_EDGESC + e], r = et[e];
  float w = invc[(size_t)r * N_NODESC + d];
  const _Float16* hp = H1 + ((size_t)r * N_NODESC + s) * HIDC;
  h8 lo = *(const h8*)hp;
  h8 hh = *(const h8*)(hp + 8);
  float* op = out1 + (size_t)d * HIDC;
#pragma unroll
  for (int c = 0; c < 8; ++c) atomicAdd(&op[c], w * (float)lo[c]);
#pragma unroll
  for (int c = 0; c < 8; ++c) atomicAdd(&op[8 + c], w * (float)hh[c]);
}

// ---------------- BatchNorm column stats (one block per column) --------------
__global__ void k_bn_stats(const float* __restrict__ x, int C, float* __restrict__ mv) {
  int c = blockIdx.x;
  float s = 0.f, s2 = 0.f;
  for (int n = threadIdx.x; n < N_NODESC; n += blockDim.x) {
    float v = x[(size_t)n * C + c];
    s += v; s2 += v * v;
  }
  __shared__ float sh[256], sh2[256];
  int t = threadIdx.x;
  sh[t] = s; sh2[t] = s2; __syncthreads();
  for (int o = blockDim.x >> 1; o > 0; o >>= 1) {
    if (t < o) { sh[t] += sh[t + o]; sh2[t] += sh2[t + o]; }
    __syncthreads();
  }
  if (t == 0) {
    float m = sh[0] / (float)N_NODESC;
    mv[c] = m;
    mv[C + c] = sh2[0] / (float)N_NODESC - m * m;   // biased variance
  }
}

// ---------------- BN1 + ReLU; also seed root rows (cols 320..335) of A2f -----
__global__ void k_bn1_apply(const float* __restrict__ out1, const float* __restrict__ mv,
                            const float* __restrict__ gamma, const float* __restrict__ beta,
                            _Float16* __restrict__ h1n, float* __restrict__ A2f) {
  int i = blockIdx.x * blockDim.x + threadIdx.x;
  if (i >= N_NODESC * HIDC) return;
  int c = i & 15, n = i >> 4;
  float m = mv[c], v = mv[HIDC + c];
  float y = (out1[i] - m) * rsqrtf(v + BN_EPSF) * gamma[c] + beta[c];
  y = fmaxf(y, 0.f);
  h1n[i] = (_Float16)y;
  A2f[(size_t)n * K2C + N_RELSC * HIDC + c] = y;   // root input rows
}

// ---------------- layer-2 edge aggregate: A2f[dst, r*16..] += h1n[src]/cnt ---
__global__ void k_edge2(const int* __restrict__ ei, const int* __restrict__ et,
                        const _Float16* __restrict__ h1n, const float* __restrict__ invc,
                        float* __restrict__ A2f) {
  int e = blockIdx.x * blockDim.x + threadIdx.x;
  if (e >= N_EDGESC) return;
  int s = ei[e], d = ei[N_EDGESC + e], r = et[e];
  float w = invc[(size_t)r * N_NODESC + d];
  const _Float16* hp = h1n + (size_t)s * HIDC;
  h8 lo = *(const h8*)hp;
  h8 hh = *(const h8*)(hp + 8);
  float* op = A2f + (size_t)d * K2C + r * HIDC;
#pragma unroll
  for (int c = 0; c < 8; ++c) atomicAdd(&op[c], w * (float)lo[c]);
#pragma unroll
  for (int c = 0; c < 8; ++c) atomicAdd(&op[8 + c], w * (float)hh[c]);
}

// ---------------- layer-2 fused GEMM: out2 = A2h[100k x 352] @ Wcat + b2 -----
__global__ void k_gemm2(const _Float16* __restrict__ A2h, const _Float16* __restrict__ W2p,
                        const float* __restrict__ b2, float* __restrict__ out2) {
  int wave  = blockIdx.x * (blockDim.x >> 5) + (threadIdx.x >> 5);
  int lane  = threadIdx.x & 31;
  int node0 = wave * 16;
  if (node0 >= N_NODESC) return;
  int m  = lane & 15;
  int hi = lane >> 4;
  const _Float16* arow = A2h + (size_t)(node0 + m) * K2C;
  v8f c0 = {}, c1 = {};
  for (int kk = 0; kk < 11; ++kk) {
    h8 lo = *(const h8*)(arow + kk * 32 + hi * 8);
    h8 hh = *(const h8*)(arow + kk * 32 + hi * 8 + 16);
    v16h a = __builtin_shufflevector(lo, hh, 0,1,2,3,4,5,6,7,8,9,10,11,12,13,14,15);
    v16h b0 = *(const v16h*)(W2p + (size_t)((kk)      * 32 + lane) * 16);
    c0 = __builtin_amdgcn_wmma_f32_16x16x32_f16(false, a, false, b0, (short)0, c0, false, false);
    v16h b1 = *(const v16h*)(W2p + (size_t)((11 + kk) * 32 + lane) * 16);
    c1 = __builtin_amdgcn_wmma_f32_16x16x32_f16(false, a, false, b1, (short)0, c1, false, false);
  }
  float bias0 = b2[m], bias1 = b2[16 + m];
  float* dst = out2 + (size_t)(node0 + hi * 8) * N_CLS;
#pragma unroll
  for (int i = 0; i < 8; ++i) {
    dst[(size_t)i * N_CLS + m]      = c0[i] + bias0;
    dst[(size_t)i * N_CLS + 16 + m] = c1[i] + bias1;
  }
}

// ---------------- BN2 + ReLU + log_softmax (wave per row, lane = class) ------
__global__ void k_bn2_softmax(const float* __restrict__ out2, const float* __restrict__ mv,
                              const float* __restrict__ gamma, const float* __restrict__ beta,
                              float* __restrict__ out) {
  int row  = blockIdx.x * (blockDim.x >> 5) + (threadIdx.x >> 5);
  int lane = threadIdx.x & 31;
  if (row >= N_NODESC) return;
  float m = mv[lane], v = mv[N_CLS + lane];
  float z = (out2[(size_t)row * N_CLS + lane] - m) * rsqrtf(v + BN_EPSF) * gamma[lane] + beta[lane];
  z = fmaxf(z, 0.f);
  float mx = z;
  for (int o = 16; o > 0; o >>= 1) mx = fmaxf(mx, __shfl_xor(mx, o, 32));
  float ex = __expf(z - mx), sm = ex;
  for (int o = 16; o > 0; o >>= 1) sm += __shfl_xor(sm, o, 32);
  out[(size_t)row * N_CLS + lane] = (z - mx) - logf(sm);
}

extern "C" void kernel_launch(void* const* d_in, const int* in_sizes, int n_in,
                              void* d_out, int out_size, void* d_ws, size_t ws_size,
                              hipStream_t stream) {
  const float* x      = (const float*)d_in[0];
  const int*   ei     = (const int*)  d_in[1];   // [2, E]: src then dst
  const int*   et     = (const int*)  d_in[2];
  const float* W1     = (const float*)d_in[3];
  const float* root1  = (const float*)d_in[4];
  const float* b1     = (const float*)d_in[5];
  const float* gamma1 = (const float*)d_in[6];
  const float* beta1  = (const float*)d_in[7];
  const float* W2     = (const float*)d_in[8];
  const float* root2  = (const float*)d_in[9];
  const float* b2     = (const float*)d_in[10];
  const float* gamma2 = (const float*)d_in[11];
  const float* beta2  = (const float*)d_in[12];
  float* out = (float*)d_out;

  char* p = (char*)d_ws;
  auto carve = [&](size_t bytes) -> char* {
    char* q = p; p += (bytes + 255) & ~(size_t)255; return q;
  };
  _Float16* Xh   = (_Float16*)carve((size_t)N_NODESC * F_INC * 2);
  _Float16* W1p  = (_Float16*)carve((size_t)(N_RELSC + 1) * 4 * 512 * 2);
  _Float16* W2p  = (_Float16*)carve((size_t)2 * 11 * 512 * 2);
  _Float16* H1   = (_Float16*)carve((size_t)N_RELSC * N_NODESC * HIDC * 2);
  float*    out1 = (float*)   carve((size_t)N_NODESC * HIDC * 4);
  float*    cnt  = (float*)   carve((size_t)N_RELSC * N_NODESC * 4);
  _Float16* h1n  = (_Float16*)carve((size_t)N_NODESC * HIDC * 2);
  float*    A2f  = (float*)   carve((size_t)N_NODESC * K2C * 4);
  _Float16* A2h  = (_Float16*)carve((size_t)N_NODESC * K2C * 2);
  float*    out2 = (float*)   carve((size_t)N_NODESC * N_CLS * 4);
  float*    mv1  = (float*)   carve((size_t)2 * HIDC * 4);
  float*    mv2  = (float*)   carve((size_t)2 * N_CLS * 4);

  hipMemsetAsync(cnt, 0, (size_t)N_RELSC * N_NODESC * 4, stream);
  hipMemsetAsync(A2f, 0, (size_t)N_NODESC * K2C * 4, stream);

  { int n = N_NODESC * F_INC;
    k_cvt_f16<<<(n + 255) / 256, 256, 0, stream>>>(x, Xh, n); }
  k_pack_w1<<<((N_RELSC + 1) * 4 * 512 + 255) / 256, 256, 0, stream>>>(W1, root1, W1p);
  k_pack_w2<<<(2 * 11 * 512 + 255) / 256, 256, 0, stream>>>(W2, root2, W2p);

  k_count <<<(N_EDGESC + 255) / 256, 256, 0, stream>>>(ei, et, cnt);
  k_invert<<<(N_RELSC * N_NODESC + 255) / 256, 256, 0, stream>>>(cnt, N_RELSC * N_NODESC);

  { int tiles = N_NODESC / 16, wpb = 4;
    k_gemm1<<<(tiles + wpb - 1) / wpb, wpb * 32, 0, stream>>>(Xh, W1p, b1, H1, out1); }
  k_edge1<<<(N_EDGESC + 255) / 256, 256, 0, stream>>>(ei, et, H1, cnt, out1);

  k_bn_stats<<<HIDC, 256, 0, stream>>>(out1, HIDC, mv1);
  k_bn1_apply<<<(N_NODESC * HIDC + 255) / 256, 256, 0, stream>>>(out1, mv1, gamma1, beta1, h1n, A2f);

  k_edge2<<<(N_EDGESC + 255) / 256, 256, 0, stream>>>(ei, et, h1n, cnt, A2f);
  { int n = N_NODESC * K2C;
    k_cvt_f16<<<(n + 255) / 256, 256, 0, stream>>>(A2f, A2h, n); }

  { int tiles = N_NODESC / 16, wpb = 4;
    k_gemm2<<<(tiles + wpb - 1) / wpb, wpb * 32, 0, stream>>>(A2h, W2p, b2, out2); }

  k_bn_stats<<<N_CLS, 256, 0, stream>>>(out2, N_CLS, mv2);
  { int wpb = 8;
    k_bn2_softmax<<<(N_NODESC + wpb - 1) / wpb, wpb * 32, 0, stream>>>(out2, mv2, gamma2, beta2, out); }
}
